// BitnetFeedForward_68564857913634
// MI455X (gfx1250) — 8-bit matrix operand order repaired and checked by reading, compiled, not run on hardware
//
#include <hip/hip_runtime.h>
#include <hip/hip_bf16.h>
#include <math.h>

typedef __attribute__((ext_vector_type(8))) int v8i;

#define TOKENS 8192
#define DIM    2048
#define INNER  8192

__device__ __forceinline__ float gelu_erf(float v) {
    return 0.5f * v * (1.0f + erff(v * 0.70710678118654752f));
}

__device__ __forceinline__ void async_ld_b128(const void* gptr, unsigned lds_off) {
    asm volatile("global_load_async_to_lds_b128 %0, %1, off"
                 :: "v"(lds_off), "v"(gptr)
                 : "memory");
}
__device__ __forceinline__ void wait_async_le1() {
    asm volatile("s_wait_asynccnt 0x1" ::: "memory");
}
__device__ __forceinline__ void wait_async_0() {
    asm volatile("s_wait_asynccnt 0x0" ::: "memory");
}

__global__ void zero_ws_kernel(float* wsum) {
    wsum[0] = 0.0f;
    wsum[1] = 0.0f;
}

__global__ __launch_bounds__(256)
void absmean_kernel(const float* __restrict__ w, long long n, float* wsum) {
    __shared__ float red[256];
    long long i = (long long)blockIdx.x * blockDim.x + threadIdx.x;
    const long long stride = (long long)gridDim.x * blockDim.x;
    float s = 0.0f;
    for (; i < n; i += stride) s += fabsf(w[i]);
    red[threadIdx.x] = s;
    __syncthreads();
    for (int o = 128; o > 0; o >>= 1) {
        if ((int)threadIdx.x < o) red[threadIdx.x] += red[threadIdx.x + o];
        __syncthreads();
    }
    if (threadIdx.x == 0) atomicAdd(wsum, red[0]);
}

__global__ __launch_bounds__(256)
void wquant_kernel(const float* __restrict__ w, signed char* __restrict__ wq,
                   long long n, const float* __restrict__ wsum) {
    const float mean  = wsum[0] / (float)n;
    const float scale = 1.0f / fmaxf(mean, 1e-5f);
    long long i = (long long)blockIdx.x * blockDim.x + threadIdx.x;
    const long long stride = (long long)gridDim.x * blockDim.x;
    for (; i < n; i += stride) {
        float q = rintf(w[i] * scale);
        q = fminf(fmaxf(q, -1.0f), 1.0f);
        wq[i] = (signed char)(int)q;
    }
}

__global__ __launch_bounds__(256)
void actquant_kernel(const float* __restrict__ x, signed char* __restrict__ xq,
                     float* __restrict__ ascale, int D) {
    __shared__ float ssum[256];
    __shared__ float smax[256];
    const int row = blockIdx.x;
    const int tid = threadIdx.x;
    const float* xr = x + (size_t)row * D;
    float ss = 0.0f, am = 0.0f;
    for (int i = tid; i < D; i += 256) {
        float v = xr[i];
        ss += v * v;
        am = fmaxf(am, fabsf(v));
    }
    ssum[tid] = ss;
    smax[tid] = am;
    __syncthreads();
    for (int o = 128; o > 0; o >>= 1) {
        if (tid < o) {
            ssum[tid] += ssum[tid + o];
            smax[tid] = fmaxf(smax[tid], smax[tid + o]);
        }
        __syncthreads();
    }
    const float norm   = fmaxf(sqrtf(ssum[0]), 1e-12f);
    const float knorm  = sqrtf((float)D) / norm;
    const float amax_n = fmaxf(smax[0] * knorm, 1e-5f);
    const float scale  = 127.0f / amax_n;
    signed char* qr = xq + (size_t)row * D;
    for (int i = tid; i < D; i += 256) {
        float q = rintf(xr[i] * knorm * scale);
        q = fminf(fmaxf(q, -128.0f), 127.0f);
        qr[i] = (signed char)(int)q;
    }
    if (tid == 0) ascale[row] = amax_n / 127.0f;
}

template <bool GELU>
__global__ __launch_bounds__(256)
void gemm_i8_kernel(const signed char* __restrict__ Aq,
                    const float* __restrict__ ascale,
                    const signed char* __restrict__ Wq,
                    const float* __restrict__ wsum, float wnumel,
                    float* __restrict__ Out,
                    int M, int N, int K) {
    __shared__ __align__(16) signed char As0[64 * 64];
    __shared__ __align__(16) signed char As1[64 * 64];

    const int tid    = threadIdx.x;
    const int lane   = tid & 31;
    const int wave   = tid >> 5;
    const int wave_m = wave >> 2;
    const int wave_n = wave & 3;
    const int blockN = blockIdx.x * 256;
    const int blockM = blockIdx.y * 64;

    const float wdq = fmaxf(wsum[0] / wnumel, 1e-5f);

    const v8i vzero = {0, 0, 0, 0, 0, 0, 0, 0};
    v8i acc[2][4];
    #pragma unroll
    for (int i = 0; i < 2; ++i)
        #pragma unroll
        for (int j = 0; j < 4; ++j) acc[i][j] = vzero;

    const int half  = lane >> 4;
    const int l15   = lane & 15;
    const int arow0 = wave_m * 32 + l15;
    const int ncol0 = blockN + wave_n * 64 + l15;

    const int ldr = tid >> 2;
    const int ldc = (tid & 3) << 4;
    const unsigned dlds0 = (unsigned)(uintptr_t)(&As0[ldr * 64 + ldc]);
    const unsigned dlds1 = (unsigned)(uintptr_t)(&As1[ldr * 64 + ldc]);
    const signed char* gsrc = Aq + (size_t)(blockM + ldr) * K + ldc;

    auto compute_step = [&](const signed char* buf, int k0) {
        v8i a[2];
        #pragma unroll
        for (int tm = 0; tm < 2; ++tm) {
            const int2* row = (const int2*)(buf + (arow0 + tm * 16) * 64);
            int2 p0 = row[2 * half + 0];
            int2 p1 = row[2 * half + 1];
            int2 p2 = row[2 * half + 4];
            int2 p3 = row[2 * half + 5];
            a[tm][0] = p0.x; a[tm][1] = p0.y; a[tm][2] = p1.x; a[tm][3] = p1.y;
            a[tm][4] = p2.x; a[tm][5] = p2.y; a[tm][6] = p3.x; a[tm][7] = p3.y;
        }
        v8i b[4];
        #pragma unroll
        for (int tn = 0; tn < 4; ++tn) {
            const int4* wr = (const int4*)(Wq + (size_t)(ncol0 + tn * 16) * K + k0);
            int4 q0 = wr[half];
            int4 q1 = wr[2 + half];
            b[tn][0] = q0.x; b[tn][1] = q0.y; b[tn][2] = q0.z; b[tn][3] = q0.w;
            b[tn][4] = q1.x; b[tn][5] = q1.y; b[tn][6] = q1.z; b[tn][7] = q1.w;
        }
        #pragma unroll
        for (int tm = 0; tm < 2; ++tm)
            #pragma unroll
            for (int tn = 0; tn < 4; ++tn)
                acc[tm][tn] = __builtin_amdgcn_wmma_i32_16x16x64_iu8(
                    true, a[tm], true, b[tn], acc[tm][tn], false, false);
    };

    const int nsteps = K >> 6;
    const int pairs  = nsteps >> 1;

    async_ld_b128(gsrc,      dlds0);
    async_ld_b128(gsrc + 64, dlds1);

    for (int p = 0; p < pairs - 1; ++p) {
        const int s = p << 1;
        wait_async_le1();
        __syncthreads();
        compute_step(As0, s << 6);
        __syncthreads();
        async_ld_b128(gsrc + (size_t)(s + 2) * 64, dlds0);

        wait_async_le1();
        __syncthreads();
        compute_step(As1, (s + 1) << 6);
        __syncthreads();
        async_ld_b128(gsrc + (size_t)(s + 3) * 64, dlds1);
    }

    wait_async_le1();
    __syncthreads();
    compute_step(As0, (nsteps - 2) << 6);
    wait_async_0();
    __syncthreads();
    compute_step(As1, (nsteps - 1) << 6);

    const int mbase = blockM + wave_m * 32 + (lane >> 4) * 8;
    #pragma unroll
    for (int tm = 0; tm < 2; ++tm) {
        #pragma unroll
        for (int r = 0; r < 8; ++r) {
            const int m  = mbase + tm * 16 + r;
            const float sa = ascale[m] * wdq;
            float* orow = Out + (size_t)m * N;
            #pragma unroll
            for (int tn = 0; tn < 4; ++tn) {
                float v = (float)acc[tm][tn][r] * sa;
                if (GELU) v = gelu_erf(v);
                orow[ncol0 + tn * 16] = v;
            }
        }
    }
}

extern "C" void kernel_launch(void* const* d_in, const int* in_sizes, int n_in,
                              void* d_out, int out_size, void* d_ws, size_t ws_size,
                              hipStream_t stream) {
    (void)in_sizes; (void)n_in; (void)out_size; (void)ws_size;

    const float* x  = (const float*)d_in[0];
    const float* w1 = (const float*)d_in[1];
    const float* w2 = (const float*)d_in[2];
    float*       out = (float*)d_out;

    char* ws = (char*)d_ws;
    size_t o = 0;
    signed char* w1q = (signed char*)(ws + o); o += (size_t)INNER * DIM;
    signed char* w2q = (signed char*)(ws + o); o += (size_t)DIM * INNER;
    signed char* xq  = (signed char*)(ws + o); o += (size_t)TOKENS * DIM;
    signed char* hq  = (signed char*)(ws + o); o += (size_t)TOKENS * INNER;
    float* h      = (float*)(ws + o); o += (size_t)TOKENS * INNER * sizeof(float);
    float* xscale = (float*)(ws + o); o += (size_t)TOKENS * sizeof(float);
    float* hscale = (float*)(ws + o); o += (size_t)TOKENS * sizeof(float);
    float* wsum   = (float*)(ws + o); o += 16;

    const long long n1 = (long long)INNER * DIM;
    const long long n2 = (long long)DIM * INNER;

    zero_ws_kernel<<<1, 1, 0, stream>>>(wsum);

    absmean_kernel<<<2048, 256, 0, stream>>>(w1, n1, wsum + 0);
    absmean_kernel<<<2048, 256, 0, stream>>>(w2, n2, wsum + 1);

    wquant_kernel<<<8192, 256, 0, stream>>>(w1, w1q, n1, wsum + 0);
    wquant_kernel<<<8192, 256, 0, stream>>>(w2, w2q, n2, wsum + 1);

    actquant_kernel<<<TOKENS, 256, 0, stream>>>(x, xq, xscale, DIM);

    gemm_i8_kernel<true><<<dim3(INNER / 256, TOKENS / 64), 256, 0, stream>>>(
        xq, xscale, w1q, wsum + 0, (float)n1, h, TOKENS, INNER, DIM);

    actquant_kernel<<<TOKENS, 256, 0, stream>>>(h, hq, hscale, INNER);

    gemm_i8_kernel<false><<<dim3(DIM / 256, TOKENS / 64), 256, 0, stream>>>(
        hq, hscale, w2q, wsum + 1, (float)n2, out, TOKENS, DIM, INNER);
}
